// QuantumNLPModel_53867479826803
// MI455X (gfx1250) — matrix operand order repaired and checked by reading, not compiled, not run on hardware
//
#include <hip/hip_runtime.h>
#include <hip/hip_bf16.h>
#include <math.h>

// ---------------------------------------------------------------------------
// CDNA5 (gfx1250) implementation of the QuantumNLP pipeline.
//   Stage 1: GEMM (bf16 WMMA 16x16x32) + bias + LayerNorm + ReLU + tanh*pi
//   Stage 2: 8-qubit statevector sim, one sample per wave32 (8 amps/lane)
//   Stage 3: <Z_i> readout (shfl_xor butterfly) + 8->4->9 MLP head
// ---------------------------------------------------------------------------

typedef __attribute__((ext_vector_type(16))) __bf16 v16bf;
typedef __attribute__((ext_vector_type(8)))  float  v8f;

#define PI_F 3.14159265358979323846f

__device__ __forceinline__ unsigned short f2bf(float f) {
  unsigned int u = __float_as_uint(f);
  u += 0x7fffu + ((u >> 16) & 1u);            // round-to-nearest-even
  return (unsigned short)(u >> 16);
}

__device__ __forceinline__ float shfx(float v, int m) {
  return __shfl_xor(v, m, 32);
}

// ---------------------------------------------------------------------------
// Kernel 0: pack W_proj [64][768] f32 into WMMA B-fragment order (bf16).
// Layout: [kstep=24][ntile=4][lane=32][elem=16], 32B per lane-chunk.
// B fragment (32x16 bf16): lane<16 -> col N=lane,    elem i<8 -> K = 32*kstep+i,   i>=8 -> K = 32*kstep+16+(i-8)
//                          lane>=16 -> col N=lane-16, elem i<8 -> K = 32*kstep+8+i, i>=8 -> K = 32*kstep+24+(i-8)
// ---------------------------------------------------------------------------
__global__ __launch_bounds__(256)
void pack_Wproj_bf16(const float* __restrict__ Wp, unsigned short* __restrict__ bp) {
  const int idx = blockIdx.x * blockDim.x + threadIdx.x;     // 49152 total
  if (idx >= 24 * 4 * 32 * 16) return;
  const int elem  = idx & 15;
  const int lane  = (idx >> 4) & 31;
  const int ntile = (idx >> 9) & 3;
  const int kstep = idx >> 11;
  const int k = kstep * 32 + ((elem & 8) ? 16 : 0) + ((lane & 16) ? 8 : 0) + (elem & 7);
  const int n = ntile * 16 + (lane & 15);
  bp[idx] = f2bf(Wp[n * 768 + k]);
}

// ---------------------------------------------------------------------------
// Kernel 1: per wave: 16 rows x 64 cols of x = A @ Wp^T + b  via WMMA,
// then LayerNorm over 64 features, ReLU, angles = tanh(x[:8]) * pi.
// 4 waves (128 threads) per block -> 64 rows per block.
// ---------------------------------------------------------------------------
__global__ __launch_bounds__(128)
void proj_ln_angles(const float* __restrict__ A,              // [B,768]
                    const unsigned short* __restrict__ Bp,    // packed bf16
                    const float* __restrict__ bproj,          // [64]
                    const float* __restrict__ lnw,            // [64]
                    const float* __restrict__ lnb,            // [64]
                    float* __restrict__ angles)               // [B,8]
{
  __shared__ float xs[4][16][65];                             // 65: bank-conflict pad
  const int lane = threadIdx.x & 31;
  const int wave = threadIdx.x >> 5;
  const int rowTile = blockIdx.x * 4 + wave;
  const int mrow = lane & 15;
  const int row  = rowTile * 16 + mrow;
  const float* arow = A + (size_t)row * 768;
  const int koff = (lane & 16) ? 8 : 0;   // A fragment K-offset (16-bit A 16x32 layout)

  v8f acc0 = {}; v8f acc1 = {}; v8f acc2 = {}; v8f acc3 = {};
  const v16bf* bfrag = (const v16bf*)Bp;

  for (int kstep = 0; kstep < 24; ++kstep) {
    const int kb = kstep * 32 + koff;
    // A fragment per lane: row = lane&15, K runs [kb..kb+7] and [kb+16..kb+23]
    float4 a0 = *(const float4*)(arow + kb + 0);
    float4 a1 = *(const float4*)(arow + kb + 4);
    float4 a2 = *(const float4*)(arow + kb + 16);
    float4 a3 = *(const float4*)(arow + kb + 20);
    if (kstep < 23) __builtin_prefetch(arow + kb + 32, 0, 3);   // global_prefetch_b8

    union { unsigned short u[16]; v16bf v; } af;
    af.u[0]  = f2bf(a0.x); af.u[1]  = f2bf(a0.y); af.u[2]  = f2bf(a0.z); af.u[3]  = f2bf(a0.w);
    af.u[4]  = f2bf(a1.x); af.u[5]  = f2bf(a1.y); af.u[6]  = f2bf(a1.z); af.u[7]  = f2bf(a1.w);
    af.u[8]  = f2bf(a2.x); af.u[9]  = f2bf(a2.y); af.u[10] = f2bf(a2.z); af.u[11] = f2bf(a2.w);
    af.u[12] = f2bf(a3.x); af.u[13] = f2bf(a3.y); af.u[14] = f2bf(a3.z); af.u[15] = f2bf(a3.w);

    const v16bf b0 = bfrag[(kstep * 4 + 0) * 32 + lane];
    const v16bf b1 = bfrag[(kstep * 4 + 1) * 32 + lane];
    const v16bf b2 = bfrag[(kstep * 4 + 2) * 32 + lane];
    const v16bf b3 = bfrag[(kstep * 4 + 3) * 32 + lane];

    acc0 = __builtin_amdgcn_wmma_f32_16x16x32_bf16(false, af.v, false, b0, (short)0, acc0, false, false);
    acc1 = __builtin_amdgcn_wmma_f32_16x16x32_bf16(false, af.v, false, b1, (short)0, acc1, false, false);
    acc2 = __builtin_amdgcn_wmma_f32_16x16x32_bf16(false, af.v, false, b2, (short)0, acc2, false, false);
    acc3 = __builtin_amdgcn_wmma_f32_16x16x32_bf16(false, af.v, false, b3, (short)0, acc3, false, false);
  }

  // C/D layout: lane<16 -> N=lane,    elem j -> M=j
  //             lane>=16 -> N=lane-16, elem j -> M=j+8
  const int nbase = lane & 15;
  const int mbase = (lane & 16) ? 8 : 0;
  const float bb0 = bproj[nbase +  0];
  const float bb1 = bproj[nbase + 16];
  const float bb2 = bproj[nbase + 32];
  const float bb3 = bproj[nbase + 48];
#pragma unroll
  for (int j = 0; j < 8; ++j) {
    xs[wave][mbase + j][nbase +  0] = acc0[j] + bb0;
    xs[wave][mbase + j][nbase + 16] = acc1[j] + bb1;
    xs[wave][mbase + j][nbase + 32] = acc2[j] + bb2;
    xs[wave][mbase + j][nbase + 48] = acc3[j] + bb3;
  }
  __syncthreads();

  // LayerNorm over 64 features: lane L handles row L&15; half-columns per lane half.
  const int cb = (lane & 16) ? 32 : 0;
  float s = 0.f, s2 = 0.f;
#pragma unroll
  for (int c = 0; c < 32; ++c) {
    const float v = xs[wave][mrow][cb + c];
    s += v; s2 += v * v;
  }
  s  += shfx(s, 16);
  s2 += shfx(s2, 16);
  const float mu   = s * (1.0f / 64.0f);
  const float var  = s2 * (1.0f / 64.0f) - mu * mu;
  const float rstd = rsqrtf(var + 1e-5f);

  if (lane < 16) {
    float o[8];
#pragma unroll
    for (int q = 0; q < 8; ++q) {
      float v = (xs[wave][mrow][q] - mu) * rstd * lnw[q] + lnb[q];
      v = fmaxf(v, 0.0f);
      o[q] = tanhf(v) * PI_F;
    }
    float4* dst = (float4*)(angles + (size_t)row * 8);
    dst[0] = make_float4(o[0], o[1], o[2], o[3]);
    dst[1] = make_float4(o[4], o[5], o[6], o[7]);
  }
}

// ---------------------------------------------------------------------------
// 8-qubit simulator helpers. State index: bits[4:0]=lane, bits[7:5]=register.
// Qubit i lives at bit position p = 7 - i (qubit 0 = MSB, matching reshape
// order of the reference).  p>=5 -> register qubit, p<5 -> lane qubit.
// ---------------------------------------------------------------------------
__device__ __forceinline__ void gate_1q(float* sr, float* si, int qubit, int lane,
                                        float ar, float ai, float br, float bi,
                                        float cr, float ci, float dr, float di) {
  const int p = 7 - qubit;
  if (p >= 5) {
    const int rm = 1 << (p - 5);
#pragma unroll
    for (int r = 0; r < 8; ++r) {
      if ((r & rm) == 0) {
        const int r1 = r | rm;
        const float x0r = sr[r],  x0i = si[r];
        const float x1r = sr[r1], x1i = si[r1];
        sr[r]  = ar * x0r - ai * x0i + br * x1r - bi * x1i;
        si[r]  = ar * x0i + ai * x0r + br * x1i + bi * x1r;
        sr[r1] = cr * x0r - ci * x0i + dr * x1r - di * x1i;
        si[r1] = cr * x0i + ci * x0r + dr * x1i + di * x1r;
      }
    }
  } else {
    const int lm = 1 << p;
    const bool hi = (lane & lm) != 0;
#pragma unroll
    for (int r = 0; r < 8; ++r) {
      const float pr = shfx(sr[r], lm), pim = shfx(si[r], lm);
      const float x0r = hi ? pr : sr[r], x0i = hi ? pim : si[r];
      const float x1r = hi ? sr[r] : pr, x1i = hi ? si[r] : pim;
      float nr, ni;
      if (hi) { nr = cr * x0r - ci * x0i + dr * x1r - di * x1i;
                ni = cr * x0i + ci * x0r + dr * x1i + di * x1r; }
      else    { nr = ar * x0r - ai * x0i + br * x1r - bi * x1i;
                ni = ar * x0i + ai * x0r + br * x1i + bi * x1r; }
      sr[r] = nr; si[r] = ni;
    }
  }
}

__device__ __forceinline__ void gate_rz(float* sr, float* si, int qubit, int lane,
                                        float c, float s) {
  const int p = 7 - qubit;   // diag(e^{-i t/2}, e^{+i t/2})
#pragma unroll
  for (int r = 0; r < 8; ++r) {
    const int idx = r * 32 + lane;
    const float ph = ((idx >> p) & 1) ? s : -s;
    const float xr = sr[r], xi = si[r];
    sr[r] = c * xr - ph * xi;
    si[r] = c * xi + ph * xr;
  }
}

__device__ __forceinline__ void cnot_g(float* sr, float* si, int ctrl, int tgt, int lane) {
  const int pc = 7 - ctrl, pt = 7 - tgt;
  if (pc >= 5 && pt >= 5) {                       // reg ctrl, reg tgt
    const int rc = 1 << (pc - 5), rt = 1 << (pt - 5);
#pragma unroll
    for (int r = 0; r < 8; ++r) {
      if ((r & rc) && !(r & rt)) {
        const int r1 = r | rt;
        const float tr = sr[r], ti = si[r];
        sr[r] = sr[r1]; si[r] = si[r1];
        sr[r1] = tr;    si[r1] = ti;
      }
    }
  } else if (pc >= 5) {                           // reg ctrl, lane tgt: full swap
    const int rc = 1 << (pc - 5), lm = 1 << pt;
#pragma unroll
    for (int r = 0; r < 8; ++r) {
      if (r & rc) { sr[r] = shfx(sr[r], lm); si[r] = shfx(si[r], lm); }
    }
  } else if (pt >= 5) {                           // lane ctrl, reg tgt
    const int lm = 1 << pc, rt = 1 << (pt - 5);
    const bool on = (lane & lm) != 0;
#pragma unroll
    for (int r = 0; r < 8; ++r) {
      if ((r & rt) == 0) {
        const int r1 = r | rt;
        const float a0r = sr[r], a0i = si[r], a1r = sr[r1], a1i = si[r1];
        sr[r]  = on ? a1r : a0r;  si[r]  = on ? a1i : a0i;
        sr[r1] = on ? a0r : a1r;  si[r1] = on ? a0i : a1i;
      }
    }
  } else {                                        // lane ctrl, lane tgt
    const int lmt = 1 << pt;
    const bool on = (lane & (1 << pc)) != 0;
#pragma unroll
    for (int r = 0; r < 8; ++r) {
      const float pr = shfx(sr[r], lmt), pim = shfx(si[r], lmt);
      sr[r] = on ? pr  : sr[r];
      si[r] = on ? pim : si[r];
    }
  }
}

// ---------------------------------------------------------------------------
// Kernel 2: one sample per wave32; 8 complex amplitudes per lane in VGPRs.
// ---------------------------------------------------------------------------
__global__ __launch_bounds__(256)
void quantum_head(const float* __restrict__ angles,    // [B,8]
                  const float* __restrict__ qw,        // [3][8][3]
                  const float* __restrict__ W1, const float* __restrict__ b1,
                  const float* __restrict__ W2, const float* __restrict__ b2,
                  float* __restrict__ out, int B)      // [B,9]
{
  const int lane = threadIdx.x & 31;
  const int s = blockIdx.x * 8 + (threadIdx.x >> 5);
  if (s >= B) return;

  // Initial product state from RY(angle) encoding: amp(b) = prod (bit? sin : cos)
  const float* a = angles + (size_t)s * 8;
  float ce[8], se[8];
#pragma unroll
  for (int i = 0; i < 8; ++i) { const float h = 0.5f * a[i]; ce[i] = cosf(h); se[i] = sinf(h); }

  float sr[8], si[8];
#pragma unroll
  for (int r = 0; r < 8; ++r) {
    const int idx = r * 32 + lane;
    float amp = 1.0f;
#pragma unroll
    for (int i = 0; i < 8; ++i) amp *= ((idx >> (7 - i)) & 1) ? se[i] : ce[i];
    sr[r] = amp; si[r] = 0.0f;
  }

  for (int l = 0; l < 3; ++l) {
#pragma unroll
    for (int i = 0; i < 8; ++i) {
      const float tx = 0.5f * qw[(l * 8 + i) * 3 + 0];
      const float ty = 0.5f * qw[(l * 8 + i) * 3 + 1];
      const float tz = 0.5f * qw[(l * 8 + i) * 3 + 2];
      const float cx = cosf(tx), sx = sinf(tx);
      const float cy = cosf(ty), sy = sinf(ty);
      const float cz = cosf(tz), sz = sinf(tz);
      gate_1q(sr, si, i, lane, cx, 0.f, 0.f, -sx, 0.f, -sx, cx, 0.f);  // RX
      gate_1q(sr, si, i, lane, cy, 0.f, -sy, 0.f, sy, 0.f, cy, 0.f);   // RY
      gate_rz(sr, si, i, lane, cz, sz);                                // RZ
    }
#pragma unroll
    for (int i = 0; i < 7; ++i) cnot_g(sr, si, i, i + 1, lane);
    cnot_g(sr, si, 0, 7, lane);                                        // ring closure
  }

  // <Z_i> = sum p(b) * (-1)^{bit_{7-i}(b)}
  float p[8], psum = 0.f;
#pragma unroll
  for (int r = 0; r < 8; ++r) { p[r] = sr[r] * sr[r] + si[r] * si[r]; psum += p[r]; }

  float ev[8];
#pragma unroll
  for (int i = 0; i < 8; ++i) {
    const int pb = 7 - i;
    if (pb >= 5) {
      const int rm = 1 << (pb - 5);
      float e = 0.f;
#pragma unroll
      for (int r = 0; r < 8; ++r) e += (r & rm) ? -p[r] : p[r];
      ev[i] = e;
    } else {
      ev[i] = (lane & (1 << pb)) ? -psum : psum;
    }
  }
#pragma unroll
  for (int i = 0; i < 8; ++i) {
    float v = ev[i];
#pragma unroll
    for (int off = 16; off >= 1; off >>= 1) v += shfx(v, off);
    ev[i] = v;
  }

  if (lane == 0) {                       // tiny MLP head 8 -> 4 -> 9
    float h[4];
#pragma unroll
    for (int j = 0; j < 4; ++j) {
      float acc = b1[j];
#pragma unroll
      for (int k = 0; k < 8; ++k) acc += W1[j * 8 + k] * ev[k];
      h[j] = fmaxf(acc, 0.f);
    }
#pragma unroll
    for (int o = 0; o < 9; ++o) {
      float acc = b2[o];
#pragma unroll
      for (int j = 0; j < 4; ++j) acc += W2[o * 4 + j] * h[j];
      out[(size_t)s * 9 + o] = acc;
    }
  }
}

// ---------------------------------------------------------------------------
extern "C" void kernel_launch(void* const* d_in, const int* in_sizes, int n_in,
                              void* d_out, int out_size, void* d_ws, size_t ws_size,
                              hipStream_t stream) {
  const float* cls = (const float*)d_in[0];   // [B,768]
  const float* Wp  = (const float*)d_in[1];   // [64,768]
  const float* bp  = (const float*)d_in[2];   // [64]
  const float* lnw = (const float*)d_in[3];   // [64]
  const float* lnb = (const float*)d_in[4];   // [64]
  const float* qw  = (const float*)d_in[5];   // [3,8,3]
  const float* W1  = (const float*)d_in[6];   // [4,8]
  const float* b1  = (const float*)d_in[7];   // [4]
  const float* W2  = (const float*)d_in[8];   // [9,4]
  const float* b2  = (const float*)d_in[9];   // [9]
  float* out = (float*)d_out;

  const int B = in_sizes[0] / 768;            // 16384

  // Workspace: [0, 96KB) packed bf16 B fragments; then B*8 floats of angles.
  unsigned short* Bpack = (unsigned short*)d_ws;
  float* angles = (float*)((char*)d_ws + (size_t)(24 * 4 * 32 * 16) * sizeof(unsigned short));

  pack_Wproj_bf16<<<192, 256, 0, stream>>>(Wp, Bpack);

  const int rowTiles = B / 16;                // 1024 wave-tiles, 4 waves/block
  proj_ln_angles<<<rowTiles / 4, 128, 0, stream>>>(cls, Bpack, bp, lnw, lnb, angles);

  quantum_head<<<(B + 7) / 8, 256, 0, stream>>>(angles, qw, W1, b1, W2, b2, out, B);
}
